// SimilarNet_8108898255115
// MI455X (gfx1250) — compile-verified, matrix operand order checked by reading, not run on hardware
//
#include <hip/hip_runtime.h>

// ---------------- problem constants (reference: B=4, N=2048, D=512, K=32, H=768)
#define B_DIM 4
#define N_DIM 2048
#define D_DIM 512
#define K_NEI 32
#define H_DIM 768
#define M_ROWS (B_DIM * N_DIM)   // 8192 token rows

// ---------------- WMMA GEMM tiling: 64 rows x 256 cols per workgroup, 8 waves.
// Wave w: row tile = w>>1 (0..3), col panel = (w&1)*128 -> 8 col tiles, 8 WMMAs/K-step.
#define BLK_ROWS 64
#define BLK_COLS 256
#define TILE_K   32
#define LDS_STRIDE 40   // f16 elems; 80B rows: 16B-aligned async stores, conflict-free reads

typedef _Float16 v16h __attribute__((ext_vector_type(16)));
typedef float    v8f  __attribute__((ext_vector_type(8)));
typedef int      v4i_vs __attribute__((vector_size(16)));   // matches async builtin param

union FragU { v16h h; unsigned u[8]; };

// K index of fragment dword i for lane-half `half` (wave32 16-bit A/B layout):
// lanes 0-15 hold K {0..7,16..23}, lanes 16-31 hold K {8..15,24..31}, packed pairs.
__device__ __forceinline__ int kmap(int i, int half) {
  int k = (i < 4) ? (2 * i) : (16 + 2 * (i - 4));
  return k + (half ? 8 : 0);
}

// pack two f32 into one dword of f16 pairs
__device__ __forceinline__ unsigned pack2(float lo, float hi) {
  union { _Float16 h[2]; unsigned u; } p;
  p.h[0] = (_Float16)lo; p.h[1] = (_Float16)hi;
  return p.u;
}

// 16-byte async global->LDS copy (ASYNCcnt-tracked, bypasses VGPRs)
__device__ __forceinline__ void async_copy16(const _Float16* g, _Float16* l) {
  __builtin_amdgcn_global_load_async_to_lds_b128(
      (__attribute__((address_space(1))) v4i_vs*)(g),
      (__attribute__((address_space(3))) v4i_vs*)(l), 0, 0);
}
// each wave issues 5 async ops per K-tile; <=5 means previous tile is resident
__device__ __forceinline__ void wait_async_le5() {
  asm volatile("s_wait_asynccnt 0x5" ::: "memory");
}

// ---------------------------------------------------------------------------
// Out = act(Ah[M,Kd] @ W + bias) with W given pre-transposed f16: WtT[Nc,Kd].
// f16 WMMA / f32 accumulate; double-buffered async LDS staging.
// grid = (Nc/BLK_COLS, M/BLK_ROWS), block = 256 (8 waves).
// ---------------------------------------------------------------------------
template <int Kd, int Nc, int RELU, int OUT32, int OUT16>
__global__ __launch_bounds__(256) void gemm_f16_kernel(
    const _Float16* __restrict__ Ah, const _Float16* __restrict__ WtT,
    const float* __restrict__ bias, float* __restrict__ Out32,
    _Float16* __restrict__ Out16) {
  __shared__ __align__(16) _Float16 As[2][BLK_ROWS * LDS_STRIDE];   // 2 x  5 KB
  __shared__ __align__(16) _Float16 Bs[2][BLK_COLS * LDS_STRIDE];   // 2 x 20 KB

  const int tid  = threadIdx.x;
  const int wave = tid >> 5;
  const int lane = tid & 31;
  const int half = lane >> 4;
  const int ln16 = lane & 15;

  const int rowBase = blockIdx.y * BLK_ROWS;
  const int colBase = blockIdx.x * BLK_COLS;
  const int rowTile = wave >> 1;              // 0..3
  const int waveCol = (wave & 1) * 128;       // 0 or 128 (8 x 16-col tiles)

  // stage one K-tile (5 async b128 ops per thread: 1 for A, 4 for B)
  auto stage = [&](int kts, _Float16* AsB, _Float16* BsB) {
    int r   = tid >> 2;
    int c16 = (tid & 3) << 3;                 // f16 elements (8 per 16B chunk)
    async_copy16(Ah + (size_t)(rowBase + r) * Kd + kts * TILE_K + c16,
                 &AsB[r * LDS_STRIDE + c16]);
#pragma unroll
    for (int it = 0; it < 4; ++it) {
      int fi = tid + (it << 8);
      int rr = fi >> 2;
      int cc = (fi & 3) << 3;
      async_copy16(WtT + (size_t)(colBase + rr) * Kd + kts * TILE_K + cc,
                   &BsB[rr * LDS_STRIDE + cc]);
    }
  };

  v8f acc[8] = {};
  constexpr int ktiles = Kd / TILE_K;

  stage(0, As[0], Bs[0]);                     // prologue: tile 0 -> buffer 0

#pragma unroll 1
  for (int kt = 0; kt < ktiles; ++kt) {
    const int cur = kt & 1;
    __syncthreads();                          // all waves done with buffer cur^1
    const int nkt = (kt + 1 < ktiles) ? (kt + 1) : 0;   // dummy refetch on last iter
    stage(nkt, As[cur ^ 1], Bs[cur ^ 1]);     // stream next tile behind compute
    wait_async_le5();                         // tile kt resident (in-order completion)
    __syncthreads();                          // cross-wave visibility

    // ---- A fragment (reused by all 8 column tiles of this wave)
    FragU a;
#pragma unroll
    for (int i = 0; i < 8; ++i)
      a.u[i] = *(const unsigned*)
          &As[cur][(rowTile * 16 + ln16) * LDS_STRIDE + kmap(i, half)];

    // ---- all 8 B fragments first (distinct registers), then 8 WMMAs back-to-back
    FragU bf[8];
#pragma unroll
    for (int t = 0; t < 8; ++t)
#pragma unroll
      for (int i = 0; i < 8; ++i)
        bf[t].u[i] = *(const unsigned*)
            &Bs[cur][(waveCol + t * 16 + ln16) * LDS_STRIDE + kmap(i, half)];
#pragma unroll
    for (int t = 0; t < 8; ++t)
      acc[t] = __builtin_amdgcn_wmma_f32_16x16x32_f16(
          false, a.h, false, bf[t].h, (short)0, acc[t], false, false);
  }

  // ---- epilogue: bias (+relu), f32 and/or f16 outputs.
  // C/D layout: lanes 0-15 N=lane, M=r; lanes 16-31 N=lane-16, M=r+8.
#pragma unroll
  for (int t = 0; t < 8; ++t) {
    int col = colBase + waveCol + t * 16 + ln16;
    float bv = bias[col];
#pragma unroll
    for (int r = 0; r < 8; ++r) {
      int m = r + (half ? 8 : 0);
      float val = acc[t][r] + bv;
      if (RELU) val = fmaxf(val, 0.0f);
      size_t off = (size_t)(rowBase + rowTile * 16 + m) * Nc + col;
      if (OUT32) Out32[off] = val;
      if (OUT16) Out16[off] = (_Float16)val;
    }
  }
}

// ---------------------------------------------------------------------------
// Weight prep: WtT[n*K + k] = f16(W[k*N + n])   (pre-transposed, f16)
// grid = (N/256, K/8), block = 256.
// ---------------------------------------------------------------------------
__global__ void transpose_f16_kernel(const float* __restrict__ W,
                                     _Float16* __restrict__ Wt, int K, int N) {
  int n  = blockIdx.x * 256 + threadIdx.x;
  int k0 = blockIdx.y * 8;
#pragma unroll
  for (int i = 0; i < 8; ++i)
    Wt[(size_t)n * K + k0 + i] = (_Float16)W[(size_t)(k0 + i) * N + n];
}

// f16 copy of a f32 tensor; one float4 -> one packed dword pair per thread.
__global__ void f32_to_f16_kernel(const float* __restrict__ src,
                                  _Float16* __restrict__ dst) {
  size_t i = (size_t)blockIdx.x * 256 + threadIdx.x;   // float4 index
  float4 v = ((const float4*)src)[i];
  unsigned* d = (unsigned*)(dst + i * 4);
  d[0] = pack2(v.x, v.y);
  d[1] = pack2(v.z, v.w);
}

// ---------------------------------------------------------------------------
// Neighbor attention: one wave per (b,n) row; writes f16 (input of Wo GEMM).
// ---------------------------------------------------------------------------
__global__ __launch_bounds__(256) void attn_kernel(
    const float* __restrict__ q, const float* __restrict__ kmat,
    const float* __restrict__ vmat, const float* __restrict__ adj,
    const int* __restrict__ inxs, _Float16* __restrict__ outh) {
  const int lane = threadIdx.x & 31;
  const int row  = blockIdx.x * 8 + (threadIdx.x >> 5);
  const int b    = row >> 11;
  const int n    = row & (N_DIM - 1);

  float4 qv[4];
#pragma unroll
  for (int c = 0; c < 4; ++c)
    qv[c] = *(const float4*)(q + (size_t)row * D_DIM + c * 128 + lane * 4);

  float sc[K_NEI];
  int   ids[K_NEI];
  const float scale = 0.044194173824159216f;   // 1/sqrt(512)

  for (int j = 0; j < K_NEI; ++j) {
    int idx = inxs[(size_t)row * K_NEI + j];
    ids[j] = idx;
    const float* kr = kmat + ((size_t)b * N_DIM + idx) * D_DIM;
    float p = 0.0f;
#pragma unroll
    for (int c = 0; c < 4; ++c) {
      float4 kv = *(const float4*)(kr + c * 128 + lane * 4);
      p += qv[c].x * kv.x + qv[c].y * kv.y + qv[c].z * kv.z + qv[c].w * kv.w;
    }
#pragma unroll
    for (int off = 16; off > 0; off >>= 1)
      p += __shfl_xor(p, off, 32);
    float a = adj[((size_t)b * N_DIM + n) * N_DIM + idx];
    sc[j] = p * scale + (a > 0.0f ? 0.0f : -1e22f);
  }

  float mx = sc[0];
#pragma unroll
  for (int j = 1; j < K_NEI; ++j) mx = fmaxf(mx, sc[j]);
  float sum = 0.0f;
#pragma unroll
  for (int j = 0; j < K_NEI; ++j) { sc[j] = __expf(sc[j] - mx); sum += sc[j]; }
  const float inv = 1.0f / sum;

  float4 acc[4];
#pragma unroll
  for (int c = 0; c < 4; ++c) { acc[c].x = 0.f; acc[c].y = 0.f; acc[c].z = 0.f; acc[c].w = 0.f; }

  for (int j = 0; j < K_NEI; ++j) {
    float w = sc[j] * inv;
    const float* vr = vmat + ((size_t)b * N_DIM + ids[j]) * D_DIM;
#pragma unroll
    for (int c = 0; c < 4; ++c) {
      float4 vv = *(const float4*)(vr + c * 128 + lane * 4);
      acc[c].x += w * vv.x; acc[c].y += w * vv.y;
      acc[c].z += w * vv.z; acc[c].w += w * vv.w;
    }
  }
#pragma unroll
  for (int c = 0; c < 4; ++c) {
    unsigned* d = (unsigned*)(outh + (size_t)row * D_DIM + c * 128 + lane * 4);
    d[0] = pack2(acc[c].x, acc[c].y);
    d[1] = pack2(acc[c].z, acc[c].w);
  }
}

// ---------------------------------------------------------------------------
// out = LayerNorm(a + (relu? max(bsrc,0) : bsrc)) * g + beta ; optional f16 copy.
// One wave per row (D=512 -> 16 f32/lane).
// ---------------------------------------------------------------------------
__global__ __launch_bounds__(256) void add_ln_kernel(
    const float* __restrict__ a, const float* __restrict__ bsrc,
    const float* __restrict__ g, const float* __restrict__ beta,
    float* __restrict__ out, _Float16* __restrict__ out16, int relu) {
  const int lane = threadIdx.x & 31;
  const int row  = blockIdx.x * 8 + (threadIdx.x >> 5);

  float4 t[4];
  float s = 0.0f, ss = 0.0f;
#pragma unroll
  for (int c = 0; c < 4; ++c) {
    size_t off = (size_t)row * D_DIM + c * 128 + lane * 4;
    float4 av = *(const float4*)(a + off);
    float4 bv = *(const float4*)(bsrc + off);
    if (relu) { bv.x = fmaxf(bv.x, 0.f); bv.y = fmaxf(bv.y, 0.f);
                bv.z = fmaxf(bv.z, 0.f); bv.w = fmaxf(bv.w, 0.f); }
    t[c].x = av.x + bv.x; t[c].y = av.y + bv.y;
    t[c].z = av.z + bv.z; t[c].w = av.w + bv.w;
    s  += t[c].x + t[c].y + t[c].z + t[c].w;
    ss += t[c].x * t[c].x + t[c].y * t[c].y + t[c].z * t[c].z + t[c].w * t[c].w;
  }
#pragma unroll
  for (int off = 16; off > 0; off >>= 1) {
    s  += __shfl_xor(s,  off, 32);
    ss += __shfl_xor(ss, off, 32);
  }
  const float mean = s * (1.0f / D_DIM);
  const float var  = ss * (1.0f / D_DIM) - mean * mean;
  const float rstd = rsqrtf(var + 1e-5f);

#pragma unroll
  for (int c = 0; c < 4; ++c) {
    int d = c * 128 + lane * 4;
    float4 gv = *(const float4*)(g + d);
    float4 bt = *(const float4*)(beta + d);
    float4 o;
    o.x = (t[c].x - mean) * rstd * gv.x + bt.x;
    o.y = (t[c].y - mean) * rstd * gv.y + bt.y;
    o.z = (t[c].z - mean) * rstd * gv.z + bt.z;
    o.w = (t[c].w - mean) * rstd * gv.w + bt.w;
    *(float4*)(out + (size_t)row * D_DIM + d) = o;
    if (out16) {
      unsigned* dh = (unsigned*)(out16 + (size_t)row * D_DIM + d);
      dh[0] = pack2(o.x, o.y);
      dh[1] = pack2(o.z, o.w);
    }
  }
}

// ---------------------------------------------------------------------------
extern "C" void kernel_launch(void* const* d_in, const int* in_sizes, int n_in,
                              void* d_out, int out_size, void* d_ws, size_t ws_size,
                              hipStream_t stream) {
  (void)in_sizes; (void)n_in; (void)out_size; (void)ws_size;
  const float* x    = (const float*)d_in[0];
  const float* adj  = (const float*)d_in[1];
  const int*   inxs = (const int*)d_in[2];
  const float* Wq = (const float*)d_in[3];  const float* bq  = (const float*)d_in[4];
  const float* Wk = (const float*)d_in[5];  const float* bk  = (const float*)d_in[6];
  const float* Wv = (const float*)d_in[7];  const float* bv  = (const float*)d_in[8];
  const float* Wo = (const float*)d_in[9];  const float* bo  = (const float*)d_in[10];
  const float* g1 = (const float*)d_in[11]; const float* be1 = (const float*)d_in[12];
  const float* W1 = (const float*)d_in[13]; const float* bf1 = (const float*)d_in[14];
  const float* W2 = (const float*)d_in[15]; const float* bf2 = (const float*)d_in[16];
  const float* g2 = (const float*)d_in[17]; const float* be2 = (const float*)d_in[18];
  float* out = (float*)d_out;

  // -------- workspace layout (bytes)
  const size_t S32 = (size_t)M_ROWS * D_DIM * 4;     // 16.8 MB per f32 activation
  const size_t S16 = (size_t)M_ROWS * D_DIM * 2;     //  8.4 MB per f16 activation
  char* p = (char*)d_ws;
  float*    qb  = (float*)p;            p += S32;    // q, later Wo-proj output
  float*    kb  = (float*)p;            p += S32;    // k, later y (f32)
  float*    vbf = (float*)p;            p += S32;    // v, later h2 (f32)
  _Float16* xh  = (_Float16*)p;         p += S16;    // f16(x)
  _Float16* ath = (_Float16*)p;         p += S16;    // f16 attention output
  _Float16* yh  = (_Float16*)p;         p += S16;    // f16(y)
  _Float16* hh  = (_Float16*)p;         p += (size_t)M_ROWS * H_DIM * 2;  // f16 FFN hidden
  _Float16* WqT = (_Float16*)p;         p += (size_t)D_DIM * D_DIM * 2;
  _Float16* WkT = (_Float16*)p;         p += (size_t)D_DIM * D_DIM * 2;
  _Float16* WvT = (_Float16*)p;         p += (size_t)D_DIM * D_DIM * 2;
  _Float16* WoT = (_Float16*)p;         p += (size_t)D_DIM * D_DIM * 2;
  _Float16* W1T = (_Float16*)p;         p += (size_t)D_DIM * H_DIM * 2;   // [768 x 512]
  _Float16* W2T = (_Float16*)p;         p += (size_t)H_DIM * D_DIM * 2;   // [512 x 768]

  dim3 blk(256);
  dim3 gD(D_DIM / BLK_COLS, M_ROWS / BLK_ROWS);      // (2, 128)
  dim3 gH(H_DIM / BLK_COLS, M_ROWS / BLK_ROWS);      // (3, 128)
  dim3 gRow(M_ROWS / 8);                             // 1024 row-wave blocks

  // -------- one-time per launch: f16 operand prep
  f32_to_f16_kernel<<<(M_ROWS * D_DIM / 4) / 256, blk, 0, stream>>>(x, xh);
  transpose_f16_kernel<<<dim3(D_DIM / 256, D_DIM / 8), blk, 0, stream>>>(Wq, WqT, D_DIM, D_DIM);
  transpose_f16_kernel<<<dim3(D_DIM / 256, D_DIM / 8), blk, 0, stream>>>(Wk, WkT, D_DIM, D_DIM);
  transpose_f16_kernel<<<dim3(D_DIM / 256, D_DIM / 8), blk, 0, stream>>>(Wv, WvT, D_DIM, D_DIM);
  transpose_f16_kernel<<<dim3(D_DIM / 256, D_DIM / 8), blk, 0, stream>>>(Wo, WoT, D_DIM, D_DIM);
  transpose_f16_kernel<<<dim3(H_DIM / 256, D_DIM / 8), blk, 0, stream>>>(W1, W1T, D_DIM, H_DIM);
  transpose_f16_kernel<<<dim3(D_DIM / 256, H_DIM / 8), blk, 0, stream>>>(W2, W2T, H_DIM, D_DIM);

  // -------- q, k, v projections (f32 outputs for the attention stage)
  gemm_f16_kernel<D_DIM, D_DIM, 0, 1, 0><<<gD, blk, 0, stream>>>(xh, WqT, bq, qb, nullptr);
  gemm_f16_kernel<D_DIM, D_DIM, 0, 1, 0><<<gD, blk, 0, stream>>>(xh, WkT, bk, kb, nullptr);
  gemm_f16_kernel<D_DIM, D_DIM, 0, 1, 0><<<gD, blk, 0, stream>>>(xh, WvT, bv, vbf, nullptr);
  // -------- masked neighbor attention (writes f16 for the Wo GEMM)
  attn_kernel<<<gRow, blk, 0, stream>>>(qb, kb, vbf, adj, inxs, ath);
  // -------- output projection, then y = LN(x + relu(o)) (f32 + f16 copies)
  gemm_f16_kernel<D_DIM, D_DIM, 0, 1, 0><<<gD, blk, 0, stream>>>(ath, WoT, bo, qb, nullptr);
  add_ln_kernel<<<gRow, blk, 0, stream>>>(x, qb, g1, be1, kb, yh, 1);
  // -------- FFN: h = relu(y@W1+b1) (f16 only); h2 = h@W2+b2; out = LN(y + h2)
  gemm_f16_kernel<D_DIM, H_DIM, 1, 0, 1><<<gH, blk, 0, stream>>>(yh, W1T, bf1, nullptr, hh);
  gemm_f16_kernel<H_DIM, D_DIM, 0, 1, 0><<<gD, blk, 0, stream>>>(hh, W2T, bf2, vbf, nullptr);
  add_ln_kernel<<<gRow, blk, 0, stream>>>(kb, vbf, g2, be2, out, nullptr, 0);
}
